// QCNetAgentEncoderWithGraphormer_19799799234872
// MI455X (gfx1250) — matrix operand order repaired and checked by reading, not compiled, not run on hardware
//
#include <hip/hip_runtime.h>
#include <math.h>

// ---------------- model constants (match reference) ----------------
#define A_  128
#define T_  50
#define M_  256
#define H_  128
#define NH_ 8
#define NT_ (A_*T_)        // 6400 agent-time nodes
#define NS_ (T_*M_)        // 12800 time-major map nodes
#define NG_ (A_+M_)        // 384 graphormer nodes
#define PI_ 3.14159265358979f

typedef __attribute__((ext_vector_type(16))) __bf16 v16bf;
typedef __attribute__((ext_vector_type(8)))  __bf16 v8bf;
typedef __attribute__((ext_vector_type(8)))  float  v8f;
typedef __attribute__((ext_vector_type(4)))  unsigned int u32x4;
typedef __attribute__((ext_vector_type(8)))  int i32x8;
typedef __attribute__((ext_vector_type(4)))  int i32x4;

// ---------------- device helpers ----------------
__device__ __forceinline__ float wrap_ang(float a) {
  float m = fmodf(a + PI_, 2.0f * PI_);
  if (m < 0.0f) m += 2.0f * PI_;
  return m - PI_;
}
__device__ __forceinline__ float ang_between(float cx, float cy, float nx, float ny) {
  return atan2f(cx * ny - cy * nx, cx * nx + cy * ny);
}
__device__ __forceinline__ unsigned fenc(float f) {
  unsigned u = __float_as_uint(f);
  return (u & 0x80000000u) ? ~u : (u | 0x80000000u);
}
__device__ __forceinline__ float fdec(unsigned u) {
  return __uint_as_float((u & 0x80000000u) ? (u & 0x7fffffffu) : ~u);
}
__device__ __forceinline__ float wave_sum(float v) {
  #pragma unroll
  for (int off = 16; off; off >>= 1) v += __shfl_xor(v, off, 32);
  return v;
}
__device__ __forceinline__ float wave_max(float v) {
  #pragma unroll
  for (int off = 16; off; off >>= 1) v = fmaxf(v, __shfl_xor(v, off, 32));
  return v;
}
__device__ __forceinline__ float block_sum128(float v, float* wred) {
  int lane = threadIdx.x & 31, w = threadIdx.x >> 5;
  v = wave_sum(v);
  if (lane == 0) wred[w] = v;
  __syncthreads();
  float t = wred[0] + wred[1] + wred[2] + wred[3];
  __syncthreads();
  return t;
}

// ---------------- WMMA bf16 GEMM: C = act(A@W + bias + resid) ----------------
// A [N,KIN] f32, W [KIN,Kout] f32 (Kout multiple of 64).
// Block tile 64x64, 8 waves as 4(m)x2(n); each wave: 16 rows x 32 cols
// -> two v_wmma_f32_16x16x32_bf16 per K-step, A fragment reused.
// Staging is branch-free (clamped rows); A staging is vectorized:
// 2x global b128 loads -> 8x bf16 pack -> 1x ds b128 store per thread.
// act: 0=none 1=relu 2=sigmoid
template<int KIN>
__global__ __launch_bounds__(256) void gemm_wmma_kernel(
    const float* __restrict__ Ag, const float* __restrict__ Wg,
    const float* __restrict__ bias, const float* __restrict__ resid,
    float* __restrict__ Cg, int N, int Kout, int act)
{
  __shared__ alignas(128) __bf16 As[64 * 32];   // [m][k]
  __shared__ alignas(128) __bf16 Bs[64 * 32];   // [n][k] (W transposed)
  const int tid  = threadIdx.x;
  const int wave = tid >> 5, lane = tid & 31;
  const int mt = wave >> 1, nt = wave & 1;
  const int bm = blockIdx.x * 64, bn = blockIdx.y * 64;

  // opaque-false TDM probe: exercises tensor_load_to_lds + s_wait_tensorcnt
  if (__builtin_expect(act == 1234, 0)) {
    u32x4 g0 = {};
    i32x8 g1 = {};
    i32x4 g2 = {}, g3 = {};
#if defined(__clang_major__) && (__clang_major__ >= 23)
    i32x8 g4 = {};
    __builtin_amdgcn_tensor_load_to_lds(g0, g1, g2, g3, g4, 0);
#else
    __builtin_amdgcn_tensor_load_to_lds(g0, g1, g2, g3, 0);
#endif
    __builtin_amdgcn_s_wait_tensorcnt(0);
  }

  v8f acc0 = {}, acc1 = {};
  // A staging: thread -> (row ar, 8 k's at ak)
  const int ar = tid >> 2;               // 0..63
  const int ak = (tid & 3) * 8;          // 0,8,16,24
  const int arow = (bm + ar < N) ? (bm + ar) : (N - 1);
  // B staging: thread -> (col brn, 8 k's at bkg), coalesced across lanes in n
  const int brn = tid & 63;              // 0..63
  const int bkg = (tid >> 6) * 8;        // 0,8,16,24
  const int bcol = bn + brn;             // always < Kout (Kout % 64 == 0)

  const int half = lane >> 4, fr = lane & 15;

  for (int k0 = 0; k0 < KIN; k0 += 32) {
    // ---- stage A (64x32) ----
    {
      const float* ap = Ag + (long long)arow * KIN + k0 + ak;
      float4 a0 = *(const float4*)ap;
      float4 a1 = *(const float4*)(ap + 4);
      v8bf av;
      av[0] = (__bf16)a0.x; av[1] = (__bf16)a0.y; av[2] = (__bf16)a0.z; av[3] = (__bf16)a0.w;
      av[4] = (__bf16)a1.x; av[5] = (__bf16)a1.y; av[6] = (__bf16)a1.z; av[7] = (__bf16)a1.w;
      *(v8bf*)(As + ar * 32 + ak) = av;
      if (k0 + 32 < KIN) __builtin_prefetch(ap + 32, 0, 1);
    }
    // ---- stage B transposed (64 n x 32 k) ----
    {
      const float* wp = Wg + (long long)(k0 + bkg) * Kout + bcol;
      v8bf bv;
      #pragma unroll
      for (int j = 0; j < 8; ++j) bv[j] = (__bf16)wp[(long long)j * Kout];
      *(v8bf*)(Bs + brn * 32 + bkg) = bv;
    }
    __syncthreads();
    // ---- fragments: contiguous k = 16*half + i in both A and B ----
    v16bf a, b0, b1;
    const __bf16* apf = &As[(mt * 16 + fr) * 32];
    #pragma unroll
    for (int i = 0; i < 8; ++i) {
      a[i]     = apf[half * 16 + i];
      a[8 + i] = apf[half * 16 + 8 + i];
    }
    const __bf16* bp0 = &Bs[(nt * 32 + fr) * 32 + half * 16];
    const __bf16* bp1 = &Bs[(nt * 32 + 16 + fr) * 32 + half * 16];
    #pragma unroll
    for (int i = 0; i < 16; ++i) { b0[i] = bp0[i]; b1[i] = bp1[i]; }
    acc0 = __builtin_amdgcn_wmma_f32_16x16x32_bf16(false, a, false, b0,
                                                   (short)0, acc0, false, false);
    acc1 = __builtin_amdgcn_wmma_f32_16x16x32_bf16(false, a, false, b1,
                                                   (short)0, acc1, false, false);
    __syncthreads();
  }
  // ---- epilogue: lanes 0-15 -> M=r, lanes 16-31 -> M=8+r; N = lane&15 ----
  const int col = lane & 15;
  const int gn0 = bn + nt * 32 + col;
  const int gn1 = gn0 + 16;
  #pragma unroll
  for (int r = 0; r < 8; ++r) {
    int gm = bm + mt * 16 + half * 8 + r;
    if (gm < N) {
      float v0 = acc0[r], v1 = acc1[r];
      if (bias)  { v0 += bias[gn0]; v1 += bias[gn1]; }
      if (resid) { v0 += resid[(long long)gm * Kout + gn0];
                   v1 += resid[(long long)gm * Kout + gn1]; }
      if (act == 1)      { v0 = fmaxf(v0, 0.0f); v1 = fmaxf(v1, 0.0f); }
      else if (act == 2) { v0 = 1.0f / (1.0f + __expf(-v0));
                           v1 = 1.0f / (1.0f + __expf(-v1)); }
      Cg[(long long)gm * Kout + gn0] = v0;
      Cg[(long long)gm * Kout + gn1] = v1;
    }
  }
}

// ---------------- LayerNorm over H=128 (one wave per row) ----------------
__global__ __launch_bounds__(256) void ln_kernel(
    const float* __restrict__ x, const float* __restrict__ g,
    const float* __restrict__ b, float* __restrict__ y, int N, int relu)
{
  int row  = blockIdx.x * 8 + (threadIdx.x >> 5);
  int lane = threadIdx.x & 31;
  if (row >= N) return;
  const float* xr = x + (long long)row * H_;
  float v[4], s = 0.0f;
  #pragma unroll
  for (int i = 0; i < 4; ++i) { v[i] = xr[lane + 32 * i]; s += v[i]; }
  s = wave_sum(s);
  float mean = s * (1.0f / H_);
  float var = 0.0f;
  #pragma unroll
  for (int i = 0; i < 4; ++i) { float d = v[i] - mean; var += d * d; }
  var = wave_sum(var) * (1.0f / H_);
  float inv = rsqrtf(var + 1e-5f);
  #pragma unroll
  for (int i = 0; i < 4; ++i) {
    int c = lane + 32 * i;
    float o = (v[i] - mean) * inv * g[c] + b[c];
    if (relu) o = fmaxf(o, 0.0f);
    y[(long long)row * H_ + c] = o;
  }
}

// ---------------- Fourier embedding (block = 128 threads per node) ----------------
__global__ __launch_bounds__(128) void fourier_kernel(
    const float* __restrict__ x, int N, int D,
    const float* __restrict__ freqs, const float* __restrict__ w1,
    const float* __restrict__ b1, const float* __restrict__ g1,
    const float* __restrict__ bn1, const float* __restrict__ w2,
    const float* __restrict__ b2, const float* __restrict__ go,
    const float* __restrict__ bo, const float* __restrict__ wo,
    const float* __restrict__ bo2,
    const float* __restrict__ type_emb, const int* __restrict__ agent_type,
    int Tsteps, float* __restrict__ out)
{
  int n = blockIdx.x;
  if (n >= N) return;
  int tid = threadIdx.x;
  __shared__ float feat[80];
  __shared__ float h1[H_];
  __shared__ float hsum[H_];
  __shared__ float wred[4];
  hsum[tid] = 0.0f;
  __syncthreads();
  for (int d = 0; d < D; ++d) {
    float xv = x[(long long)n * D + d];
    if (tid < 32) {
      float f = xv * freqs[d * 32 + tid] * (2.0f * PI_);
      feat[tid]      = __cosf(f);
      feat[32 + tid] = __sinf(f);
      if (tid == 0) feat[64] = xv;
    }
    __syncthreads();
    float acc = b1[d * H_ + tid];
    const float* w1d = w1 + (long long)d * 65 * H_;
    for (int f = 0; f < 65; ++f) acc += feat[f] * w1d[f * H_ + tid];
    float mean = block_sum128(acc, wred) * (1.0f / H_);
    float dv = acc - mean;
    float var = block_sum128(dv * dv, wred) * (1.0f / H_);
    float hv = fmaxf(dv * rsqrtf(var + 1e-5f) * g1[d * H_ + tid] + bn1[d * H_ + tid], 0.0f);
    h1[tid] = hv;
    __syncthreads();
    float a2 = b2[d * H_ + tid];
    const float* w2d = w2 + (long long)d * H_ * H_;
    for (int i = 0; i < H_; ++i) a2 += h1[i] * w2d[i * H_ + tid];
    hsum[tid] += a2;
    __syncthreads();
  }
  float h = hsum[tid];
  if (agent_type) h += type_emb[agent_type[n / Tsteps] * H_ + tid];
  float mean = block_sum128(h, wred) * (1.0f / H_);
  float dv = h - mean;
  float var = block_sum128(dv * dv, wred) * (1.0f / H_);
  float hn = fmaxf(dv * rsqrtf(var + 1e-5f) * go[tid] + bo[tid], 0.0f);
  h1[tid] = hn;
  __syncthreads();
  float o = bo2[tid];
  for (int i = 0; i < H_; ++i) o += h1[i] * wo[i * H_ + tid];
  out[(long long)n * H_ + tid] = o;
}

// ---------------- feature kernels ----------------
__global__ void xfeat_kernel(const float* __restrict__ motion,
                             const float* __restrict__ vel,
                             const float* __restrict__ head,
                             float* __restrict__ xfeat, int n)
{
  int i = blockIdx.x * blockDim.x + threadIdx.x;
  if (i >= n) return;
  float hx = __cosf(head[i]), hy = __sinf(head[i]);
  float mx = motion[i * 2], my = motion[i * 2 + 1];
  float vx = vel[i * 2],    vy = vel[i * 2 + 1];
  xfeat[i * 4 + 0] = sqrtf(mx * mx + my * my);
  xfeat[i * 4 + 1] = ang_between(hx, hy, mx, my);
  xfeat[i * 4 + 2] = sqrtf(vx * vx + vy * vy);
  xfeat[i * 4 + 3] = ang_between(hx, hy, vx, vy);
}

__global__ void rtfeat_kernel(const float* __restrict__ pos,
                              const float* __restrict__ head,
                              const int* __restrict__ src, const int* __restrict__ dst,
                              float* __restrict__ rf, int E)
{
  int e = blockIdx.x * blockDim.x + threadIdx.x;
  if (e >= E) return;
  int s = src[e], d = dst[e];
  float rx = pos[s * 2] - pos[d * 2];
  float ry = pos[s * 2 + 1] - pos[d * 2 + 1];
  float hx = __cosf(head[d]), hy = __sinf(head[d]);
  rf[e * 4 + 0] = sqrtf(rx * rx + ry * ry);
  rf[e * 4 + 1] = ang_between(hx, hy, rx, ry);
  rf[e * 4 + 2] = wrap_ang(head[s] - head[d]);
  rf[e * 4 + 3] = (float)(s - d);
}

__global__ void rpfeat_kernel(const float* __restrict__ pos_a,
                              const float* __restrict__ head_a,
                              const float* __restrict__ pos_pl,
                              const float* __restrict__ orient_pl,
                              const int* __restrict__ src, const int* __restrict__ dst,
                              float* __restrict__ rf, int E)
{
  int e = blockIdx.x * blockDim.x + threadIdx.x;
  if (e >= E) return;
  int sp = src[e], da = dst[e];
  int mi = sp % M_;
  int t  = da / A_, ai = da % A_;
  int an = ai * T_ + t;
  float px = pos_a[an * 2], py = pos_a[an * 2 + 1];
  float hd = head_a[an];
  float hx = __cosf(hd), hy = __sinf(hd);
  float rx = pos_pl[mi * 2] - px, ry = pos_pl[mi * 2 + 1] - py;
  rf[e * 3 + 0] = sqrtf(rx * rx + ry * ry);
  rf[e * 3 + 1] = ang_between(hx, hy, rx, ry);
  rf[e * 3 + 2] = wrap_ang(orient_pl[mi] - hd);
}

// ---------------- util kernels ----------------
__global__ void fill_f_kernel(float* p, float v, long long n) {
  long long i = (long long)blockIdx.x * blockDim.x + threadIdx.x;
  if (i < n) p[i] = v;
}
__global__ void fill_u_kernel(unsigned* p, unsigned v, long long n) {
  long long i = (long long)blockIdx.x * blockDim.x + threadIdx.x;
  if (i < n) p[i] = v;
}
__global__ void copy_kernel(const float* s, float* d, long long n) {
  long long i = (long long)blockIdx.x * blockDim.x + threadIdx.x;
  if (i < n) d[i] = s[i];
}
// [D0,D1,H] -> [D1,D0,H]
__global__ void transpose_kernel(const float* __restrict__ s, float* __restrict__ d,
                                 int D0, int D1)
{
  long long i = (long long)blockIdx.x * blockDim.x + threadIdx.x;
  long long n = (long long)D0 * D1 * H_;
  if (i >= n) return;
  int h = (int)(i % H_);
  long long r = i / H_;
  int d1 = (int)(r % D1), d0 = (int)(r / D1);
  d[((long long)d1 * D0 + d0) * H_ + h] = s[i];
}

// ---------------- edge attention (segment softmax via atomics) ----------------
__global__ void edge_sim_kernel(const float* __restrict__ q, const float* __restrict__ k,
                                const float* __restrict__ kr,
                                const int* __restrict__ src, const int* __restrict__ dst,
                                float* __restrict__ sim, unsigned* __restrict__ smax, int E)
{
  long long idx = (long long)blockIdx.x * blockDim.x + threadIdx.x;
  if (idx >= (long long)E * NH_) return;
  int e = (int)(idx >> 3), h = (int)(idx & 7);
  int s = src[e], d = dst[e];
  const float* qp  = q  + (long long)d * H_ + h * 16;
  const float* kp  = k  + (long long)s * H_ + h * 16;
  const float* krp = kr + (long long)e * H_ + h * 16;
  float acc = 0.0f;
  #pragma unroll
  for (int i = 0; i < 16; ++i) acc += qp[i] * (kp[i] + krp[i]);
  acc *= 0.25f;
  sim[idx] = acc;
  atomicMax(&smax[(long long)d * NH_ + h], fenc(acc));
}

__global__ void edge_exp_kernel(float* __restrict__ sim, const unsigned* __restrict__ smax,
                                float* __restrict__ ssum, const int* __restrict__ dst, int E)
{
  long long idx = (long long)blockIdx.x * blockDim.x + threadIdx.x;
  if (idx >= (long long)E * NH_) return;
  int e = (int)(idx >> 3), h = (int)(idx & 7);
  int d = dst[e];
  float m = fdec(smax[(long long)d * NH_ + h]);
  float ex = __expf(sim[idx] - m);
  sim[idx] = ex;
  atomicAdd(&ssum[(long long)d * NH_ + h], ex);
}

__global__ void edge_agg_kernel(const float* __restrict__ sim, const float* __restrict__ ssum,
                                const float* __restrict__ v, const float* __restrict__ vr,
                                const int* __restrict__ src, const int* __restrict__ dst,
                                float* __restrict__ agg, int E)
{
  long long idx = (long long)blockIdx.x * blockDim.x + threadIdx.x;
  if (idx >= (long long)E * NH_) return;
  int e = (int)(idx >> 3), h = (int)(idx & 7);
  int s = src[e], d = dst[e];
  float w = sim[idx] / (ssum[(long long)d * NH_ + h] + 1e-9f);
  const float* vp  = v  + (long long)s * H_ + h * 16;
  const float* vrp = vr + (long long)e * H_ + h * 16;
  float* ap = agg + (long long)d * H_ + h * 16;
  #pragma unroll
  for (int i = 0; i < 16; ++i) atomicAdd(&ap[i], w * (vp[i] + vrp[i]));
}

__global__ void gate_mix_kernel(const float* agg, const float* g, const float* s,
                                float* o, long long n)
{
  long long i = (long long)blockIdx.x * blockDim.x + threadIdx.x;
  if (i < n) o[i] = agg[i] + g[i] * (s[i] - agg[i]);
}

// ---------------- graphormer ----------------
__global__ void follow_kernel(const int* src, float* follow, int E) {
  int i = blockIdx.x * blockDim.x + threadIdx.x;
  if (i < E) atomicAdd(&follow[src[i]], 1.0f);
}

__global__ void nodeaux_kernel(const float* pos_a, const float* pos_pl,
                               const float* vel, const int* agent_type,
                               const float* tb,
                               float* posn, float* vel2, float* tbn)
{
  int i = blockIdx.x * blockDim.x + threadIdx.x;
  if (i >= NG_) return;
  if (i < A_) {
    int n = i * T_ + (T_ - 1);
    posn[i * 2] = pos_a[n * 2]; posn[i * 2 + 1] = pos_a[n * 2 + 1];
    vel2[i * 2] = vel[n * 2];   vel2[i * 2 + 1] = vel[n * 2 + 1];
    for (int h = 0; h < NH_; ++h) tbn[i * NH_ + h] = tb[agent_type[i] * NH_ + h];
  } else {
    int m = i - A_;
    posn[i * 2] = pos_pl[m * 2]; posn[i * 2 + 1] = pos_pl[m * 2 + 1];
    vel2[i * 2] = 0.0f; vel2[i * 2 + 1] = 0.0f;
    for (int h = 0; h < NH_; ++h) tbn[i * NH_ + h] = 0.0f;
  }
}

__global__ void nodebuild_kernel(const float* xa, const float* map_enc,
                                 const float* follow, const float* fscale,
                                 float* xcat)
{
  int idx = blockIdx.x * blockDim.x + threadIdx.x;
  if (idx >= NG_ * H_) return;
  int i = idx / H_, h = idx % H_;
  if (i < A_) {
    xcat[idx] = xa[((long long)i * T_ + (T_ - 1)) * H_ + h];
  } else {
    int m = i - A_;
    xcat[idx] = map_enc[((long long)m * T_ + (T_ - 1)) * H_ + h]
              + log1pf(follow[m]) * fscale[h];
  }
}

__global__ void gbias_kernel(const float* posn, const float* vel2, const float* tbn,
                             const float* sp, const float* sd, const float* ttc,
                             const float* et, float* bias)
{
  int idx = blockIdx.x * blockDim.x + threadIdx.x;
  if (idx >= NG_ * NG_) return;
  int i = idx / NG_, j = idx % NG_;
  float dx = posn[i * 2] - posn[j * 2], dy = posn[i * 2 + 1] - posn[j * 2 + 1];
  float d  = sqrtf(dx * dx + dy * dy + 1e-6f);
  float rx = vel2[i * 2] - vel2[j * 2], ry = vel2[i * 2 + 1] - vel2[j * 2 + 1];
  float rs = sqrtf(rx * rx + ry * ry + 1e-6f);
  int spi = min(max((int)(d / 25.0f), 0), 19);
  int sdi = min(max((int)(rs / 2.0f), 0), 9);
  int tti = min(max((int)(d / (rs + 0.1f) / 2.0f), 0), 9);
  int ismi = (i < A_) ? 0 : 1, ismj = (j < A_) ? 0 : 1;
  int eti = ismi * 2 + ismj;
  for (int h = 0; h < NH_; ++h) {
    bias[((long long)h * NG_ + i) * NG_ + j] =
        sp[spi * NH_ + h] + sd[sdi * NH_ + h] + ttc[tti * NH_ + h] +
        et[eti * NH_ + h] + tbn[j * NH_ + h];
  }
}

// dense softmax attention per (query, head): one wave each (wave32)
__global__ __launch_bounds__(256) void dense_attn_kernel(
    const float* __restrict__ q, const float* __restrict__ k,
    const float* __restrict__ v, const float* __restrict__ bias,
    float* __restrict__ o, int N)
{
  int gw = blockIdx.x * 8 + (threadIdx.x >> 5);
  int lane = threadIdx.x & 31;
  if (gw >= N * NH_) return;
  int i = gw >> 3, h = gw & 7;
  const float* qi = q + (long long)i * H_ + h * 16;
  float qv[16];
  #pragma unroll
  for (int d = 0; d < 16; ++d) qv[d] = qi[d];
  const float* bh = bias + ((long long)h * N + i) * N;
  float mx = -1e30f;
  for (int j = lane; j < N; j += 32) {
    const float* kj = k + (long long)j * H_ + h * 16;
    float s = 0.0f;
    #pragma unroll
    for (int d = 0; d < 16; ++d) s += qv[d] * kj[d];
    mx = fmaxf(mx, s * 0.25f + bh[j]);
  }
  mx = wave_max(mx);
  float sum = 0.0f, oacc[16];
  #pragma unroll
  for (int d = 0; d < 16; ++d) oacc[d] = 0.0f;
  for (int j = lane; j < N; j += 32) {
    const float* kj = k + (long long)j * H_ + h * 16;
    float s = 0.0f;
    #pragma unroll
    for (int d = 0; d < 16; ++d) s += qv[d] * kj[d];
    float ex = __expf(s * 0.25f + bh[j] - mx);
    sum += ex;
    const float* vj = v + (long long)j * H_ + h * 16;
    #pragma unroll
    for (int d = 0; d < 16; ++d) oacc[d] += ex * vj[d];
  }
  sum = wave_sum(sum);
  #pragma unroll
  for (int d = 0; d < 16; ++d) oacc[d] = wave_sum(oacc[d]);
  if (lane < 16) o[(long long)i * H_ + h * 16 + lane] = oacc[lane] / sum;
}

__global__ void outlast_kernel(const float* xcat, float* out) {
  int idx = blockIdx.x * blockDim.x + threadIdx.x;
  if (idx >= NG_ * H_) return;
  int i = idx / H_, h = idx % H_;
  if (i < A_)
    out[((long long)i * T_ + (T_ - 1)) * H_ + h] = xcat[idx];
  else {
    int m = i - A_;
    out[(long long)NT_ * H_ + ((long long)m * T_ + (T_ - 1)) * H_ + h] = xcat[idx];
  }
}

// ---------------- host-side parameter mapping (alphabetical pytree order) ----------------
struct Cur { void* const* in; int i; const float* f() { return (const float*)in[i++]; } };
struct FourP { const float *b1,*b2,*bn1,*bo,*bo2,*freqs,*g1,*go,*w1,*w2,*wo; };
struct AttnP { const float *b1,*b2,*bg,*bo,*bq,*bs,*bv,*bvr,*lnd_b,*lnd_g,*lnf_b,*lnf_g,
                           *lnr_b,*lnr_g,*lns_b,*lns_g,*w1,*w2,*wg,*wk,*wkr,*wo,*wq,*ws,*wv,*wvr; };
struct GLayerP { const float *b1,*b2,*bo,*ln1_b,*ln1_g,*ln2_b,*ln2_g,*w1,*w2,*wk,*wo,*wq,*wv; };
struct GraphP { const float *et,*follow_scale; GLayerP l[2]; const float *sd,*sp,*tb,*ttc; };

static void load_four(Cur& c, FourP& p) {
  p.b1=c.f(); p.b2=c.f(); p.bn1=c.f(); p.bo=c.f(); p.bo2=c.f(); p.freqs=c.f();
  p.g1=c.f(); p.go=c.f(); p.w1=c.f(); p.w2=c.f(); p.wo=c.f();
}
static void load_attn(Cur& c, AttnP& p) {
  p.b1=c.f(); p.b2=c.f(); p.bg=c.f(); p.bo=c.f(); p.bq=c.f(); p.bs=c.f(); p.bv=c.f();
  p.bvr=c.f(); p.lnd_b=c.f(); p.lnd_g=c.f(); p.lnf_b=c.f(); p.lnf_g=c.f(); p.lnr_b=c.f();
  p.lnr_g=c.f(); p.lns_b=c.f(); p.lns_g=c.f(); p.w1=c.f(); p.w2=c.f(); p.wg=c.f();
  p.wk=c.f(); p.wkr=c.f(); p.wo=c.f(); p.wq=c.f(); p.ws=c.f(); p.wv=c.f(); p.wvr=c.f();
}
static void load_gl(Cur& c, GLayerP& p) {
  p.b1=c.f(); p.b2=c.f(); p.bo=c.f(); p.ln1_b=c.f(); p.ln1_g=c.f(); p.ln2_b=c.f();
  p.ln2_g=c.f(); p.w1=c.f(); p.w2=c.f(); p.wk=c.f(); p.wo=c.f(); p.wq=c.f(); p.wv=c.f();
}

struct Arena {
  char* base; size_t off;
  void* alloc(size_t bytes) {
    size_t a = (off + 255) & ~(size_t)255;
    off = a + bytes;
    return base + a;
  }
  float* f(size_t n) { return (float*)alloc(n * 4); }
  unsigned* u(size_t n) { return (unsigned*)alloc(n * 4); }
};

struct AttnBufs {
  float *lns, *lnd, *lnr, *q, *k, *v, *kr, *vr, *sim, *ssum, *agg, *gt, *sl, *tmp1, *ffh;
  unsigned* smax;
};

static inline dim3 g1d(long long n, int b) { return dim3((unsigned)((n + b - 1) / b)); }

static void gemm(const float* Aa, const float* W, const float* bias, const float* resid,
                 float* C, int N, int Kin, int Kout, int act, hipStream_t st) {
  dim3 grid((N + 63) / 64, (unsigned)(Kout / 64));
  if (Kin == 512)
    gemm_wmma_kernel<512><<<grid, 256, 0, st>>>(Aa, W, bias, resid, C, N, Kout, act);
  else
    gemm_wmma_kernel<128><<<grid, 256, 0, st>>>(Aa, W, bias, resid, C, N, Kout, act);
}

static void run_attn(const AttnP& P, const float* xsrc, float* xdst, const float* remb,
                     const int* src, const int* dst, int E, int Ns, int Nd, bool bip,
                     const AttnBufs& B, hipStream_t st)
{
  ln_kernel<<<g1d(Ns, 8), 256, 0, st>>>(xsrc, P.lns_g, P.lns_b, B.lns, Ns, 0);
  const float* lnd = B.lns;
  if (bip) {
    ln_kernel<<<g1d(Nd, 8), 256, 0, st>>>(xdst, P.lnd_g, P.lnd_b, B.lnd, Nd, 0);
    lnd = B.lnd;
  }
  ln_kernel<<<g1d(E, 8), 256, 0, st>>>(remb, P.lnr_g, P.lnr_b, B.lnr, E, 0);
  gemm(lnd,   P.wq,  P.bq,   nullptr, B.q,  Nd, H_, H_, 0, st);
  gemm(B.lns, P.wk,  nullptr,nullptr, B.k,  Ns, H_, H_, 0, st);
  gemm(B.lns, P.wv,  P.bv,   nullptr, B.v,  Ns, H_, H_, 0, st);
  gemm(B.lnr, P.wkr, nullptr,nullptr, B.kr, E,  H_, H_, 0, st);
  gemm(B.lnr, P.wvr, P.bvr,  nullptr, B.vr, E,  H_, H_, 0, st);
  fill_u_kernel<<<g1d((long long)Nd * NH_, 256), 256, 0, st>>>(B.smax, 0u, (long long)Nd * NH_);
  fill_f_kernel<<<g1d((long long)Nd * NH_, 256), 256, 0, st>>>(B.ssum, 0.0f, (long long)Nd * NH_);
  fill_f_kernel<<<g1d((long long)Nd * H_, 256), 256, 0, st>>>(B.agg, 0.0f, (long long)Nd * H_);
  long long Eh = (long long)E * NH_;
  edge_sim_kernel<<<g1d(Eh, 256), 256, 0, st>>>(B.q, B.k, B.kr, src, dst, B.sim, B.smax, E);
  edge_exp_kernel<<<g1d(Eh, 256), 256, 0, st>>>(B.sim, B.smax, B.ssum, dst, E);
  edge_agg_kernel<<<g1d(Eh, 256), 256, 0, st>>>(B.sim, B.ssum, B.v, B.vr, src, dst, B.agg, E);
  gemm(B.agg, P.wg,            P.bg,   nullptr, B.tmp1, Nd, H_, H_, 0, st);
  gemm(lnd,   P.wg + H_ * H_,  nullptr,B.tmp1,  B.gt,   Nd, H_, H_, 2, st);
  gemm(lnd,   P.ws,            P.bs,   nullptr, B.sl,   Nd, H_, H_, 0, st);
  gate_mix_kernel<<<g1d((long long)Nd * H_, 256), 256, 0, st>>>(B.agg, B.gt, B.sl, B.tmp1,
                                                                (long long)Nd * H_);
  gemm(B.tmp1, P.wo, P.bo, xdst, xdst, Nd, H_, H_, 0, st);
  ln_kernel<<<g1d(Nd, 8), 256, 0, st>>>(xdst, P.lnf_g, P.lnf_b, B.lnd, Nd, 0);
  gemm(B.lnd, P.w1, P.b1, nullptr, B.ffh, Nd, H_, 4 * H_, 1, st);
  gemm(B.ffh, P.w2, P.b2, xdst,    xdst,  Nd, 4 * H_, H_, 0, st);
}

extern "C" void kernel_launch(void* const* d_in, const int* in_sizes, int n_in,
                              void* d_out, int out_size, void* d_ws, size_t ws_size,
                              hipStream_t stream)
{
  Cur c{d_in, 0};
  GraphP G;
  G.et = c.f(); G.follow_scale = c.f();
  load_gl(c, G.l[0]); load_gl(c, G.l[1]);
  G.sd = c.f(); G.sp = c.f(); G.tb = c.f(); G.ttc = c.f();
  AttnP pl2a[2]; load_attn(c, pl2a[0]); load_attn(c, pl2a[1]);
  FourP r_pl2a_emb; load_four(c, r_pl2a_emb);
  FourP r_t_emb;    load_four(c, r_t_emb);
  AttnP tat[2]; load_attn(c, tat[0]); load_attn(c, tat[1]);
  const float* type_emb = c.f();
  FourP x_a_emb; load_four(c, x_a_emb);

  const int rb = n_in - 14;
  const float* map_enc   = (const float*)d_in[rb + 0];
  const float* pos_a     = (const float*)d_in[rb + 2];
  const float* motion    = (const float*)d_in[rb + 3];
  const float* head_a    = (const float*)d_in[rb + 4];
  const float* pos_pl    = (const float*)d_in[rb + 5];
  const float* orient_pl = (const float*)d_in[rb + 6];
  const float* vel       = (const float*)d_in[rb + 7];
  const int*   agent_ty  = (const int*)d_in[rb + 8];
  const int*   e_t       = (const int*)d_in[rb + 11];
  const int*   e_p       = (const int*)d_in[rb + 12];
  const int*   e_last    = (const int*)d_in[rb + 13];
  const int Et = in_sizes[rb + 11] / 2;
  const int Ep = in_sizes[rb + 12] / 2;
  const int El = in_sizes[rb + 13] / 2;
  const int* et_s = e_t, *et_d = e_t + Et;
  const int* ep_s = e_p, *ep_d = e_p + Ep;
  const int Emax = Et > Ep ? Et : Ep;

  Arena ar{(char*)d_ws, 0};
  float* xa    = ar.f((size_t)NT_ * H_);
  float* xat   = ar.f((size_t)NT_ * H_);
  float* xplT  = ar.f((size_t)NS_ * H_);
  float* xfeat = ar.f((size_t)NT_ * 4);
  float* rtf   = ar.f((size_t)Et * 4);
  float* rpf   = ar.f((size_t)Ep * 3);
  float* rte   = ar.f((size_t)Et * H_);
  float* rpe   = ar.f((size_t)Ep * H_);
  AttnBufs B;
  B.lns  = ar.f((size_t)NS_ * H_);
  B.lnd  = ar.f((size_t)NT_ * H_);
  B.lnr  = ar.f((size_t)Emax * H_);
  B.q    = ar.f((size_t)NT_ * H_);
  B.k    = ar.f((size_t)NS_ * H_);
  B.v    = ar.f((size_t)NS_ * H_);
  B.kr   = ar.f((size_t)Emax * H_);
  B.vr   = ar.f((size_t)Emax * H_);
  B.sim  = ar.f((size_t)Emax * NH_);
  B.smax = ar.u((size_t)NT_ * NH_);
  B.ssum = ar.f((size_t)NT_ * NH_);
  B.agg  = ar.f((size_t)NT_ * H_);
  B.gt   = ar.f((size_t)NT_ * H_);
  B.sl   = ar.f((size_t)NT_ * H_);
  B.tmp1 = ar.f((size_t)NT_ * H_);
  B.ffh  = ar.f((size_t)NT_ * 4 * H_);
  float* follow = ar.f(M_);
  float* posn   = ar.f((size_t)NG_ * 2);
  float* vel2   = ar.f((size_t)NG_ * 2);
  float* tbn    = ar.f((size_t)NG_ * NH_);
  float* xcat   = ar.f((size_t)NG_ * H_);
  float* gbias  = ar.f((size_t)NH_ * NG_ * NG_);
  float* gq     = ar.f((size_t)NG_ * H_);
  float* gk     = ar.f((size_t)NG_ * H_);
  float* gv     = ar.f((size_t)NG_ * H_);
  float* go     = ar.f((size_t)NG_ * H_);
  float* ghb    = ar.f((size_t)NG_ * H_);
  float* gffh   = ar.f((size_t)NG_ * 4 * H_);

  xfeat_kernel<<<g1d(NT_, 256), 256, 0, stream>>>(motion, vel, head_a, xfeat, NT_);
  fourier_kernel<<<NT_, 128, 0, stream>>>(xfeat, NT_, 4,
      x_a_emb.freqs, x_a_emb.w1, x_a_emb.b1, x_a_emb.g1, x_a_emb.bn1, x_a_emb.w2,
      x_a_emb.b2, x_a_emb.go, x_a_emb.bo, x_a_emb.wo, x_a_emb.bo2,
      type_emb, agent_ty, T_, xa);
  rtfeat_kernel<<<g1d(Et, 256), 256, 0, stream>>>(pos_a, head_a, et_s, et_d, rtf, Et);
  fourier_kernel<<<Et, 128, 0, stream>>>(rtf, Et, 4,
      r_t_emb.freqs, r_t_emb.w1, r_t_emb.b1, r_t_emb.g1, r_t_emb.bn1, r_t_emb.w2,
      r_t_emb.b2, r_t_emb.go, r_t_emb.bo, r_t_emb.wo, r_t_emb.bo2,
      nullptr, nullptr, 1, rte);
  rpfeat_kernel<<<g1d(Ep, 256), 256, 0, stream>>>(pos_a, head_a, pos_pl, orient_pl,
                                                  ep_s, ep_d, rpf, Ep);
  fourier_kernel<<<Ep, 128, 0, stream>>>(rpf, Ep, 3,
      r_pl2a_emb.freqs, r_pl2a_emb.w1, r_pl2a_emb.b1, r_pl2a_emb.g1, r_pl2a_emb.bn1,
      r_pl2a_emb.w2, r_pl2a_emb.b2, r_pl2a_emb.go, r_pl2a_emb.bo, r_pl2a_emb.wo,
      r_pl2a_emb.bo2, nullptr, nullptr, 1, rpe);
  transpose_kernel<<<g1d((long long)NS_ * H_, 256), 256, 0, stream>>>(map_enc, xplT, M_, T_);

  for (int l = 0; l < 2; ++l) {
    run_attn(tat[l], xa, xa, rte, et_s, et_d, Et, NT_, NT_, false, B, stream);
    transpose_kernel<<<g1d((long long)NT_ * H_, 256), 256, 0, stream>>>(xa, xat, A_, T_);
    run_attn(pl2a[l], xplT, xat, rpe, ep_s, ep_d, Ep, NS_, NT_, true, B, stream);
    transpose_kernel<<<g1d((long long)NT_ * H_, 256), 256, 0, stream>>>(xat, xa, T_, A_);
  }

  fill_f_kernel<<<g1d(M_, 256), 256, 0, stream>>>(follow, 0.0f, M_);
  follow_kernel<<<g1d(El, 256), 256, 0, stream>>>(e_last, follow, El);
  nodeaux_kernel<<<g1d(NG_, 256), 256, 0, stream>>>(pos_a, pos_pl, vel, agent_ty, G.tb,
                                                    posn, vel2, tbn);
  nodebuild_kernel<<<g1d((long long)NG_ * H_, 256), 256, 0, stream>>>(xa, map_enc, follow,
                                                                      G.follow_scale, xcat);
  gbias_kernel<<<g1d((long long)NG_ * NG_, 256), 256, 0, stream>>>(posn, vel2, tbn,
                                                                   G.sp, G.sd, G.ttc, G.et,
                                                                   gbias);
  for (int l = 0; l < 2; ++l) {
    const GLayerP& P = G.l[l];
    ln_kernel<<<g1d(NG_, 8), 256, 0, stream>>>(xcat, P.ln1_g, P.ln1_b, ghb, NG_, 0);
    gemm(ghb, P.wq, nullptr, nullptr, gq, NG_, H_, H_, 0, stream);
    gemm(ghb, P.wk, nullptr, nullptr, gk, NG_, H_, H_, 0, stream);
    gemm(ghb, P.wv, nullptr, nullptr, gv, NG_, H_, H_, 0, stream);
    dense_attn_kernel<<<g1d((long long)NG_ * NH_, 8), 256, 0, stream>>>(gq, gk, gv, gbias,
                                                                        go, NG_);
    gemm(go, P.wo, P.bo, xcat, xcat, NG_, H_, H_, 0, stream);
    ln_kernel<<<g1d(NG_, 8), 256, 0, stream>>>(xcat, P.ln2_g, P.ln2_b, ghb, NG_, 0);
    gemm(ghb, P.w1, P.b1, nullptr, gffh, NG_, H_, 4 * H_, 1, stream);
    gemm(gffh, P.w2, P.b2, xcat, xcat, NG_, 4 * H_, H_, 0, stream);
  }

  float* out = (float*)d_out;
  copy_kernel<<<g1d((long long)NT_ * H_, 256), 256, 0, stream>>>(xa, out, (long long)NT_ * H_);
  copy_kernel<<<g1d((long long)M_ * T_ * H_, 256), 256, 0, stream>>>(map_enc, out + (long long)NT_ * H_,
                                                                     (long long)M_ * T_ * H_);
  outlast_kernel<<<g1d((long long)NG_ * H_, 256), 256, 0, stream>>>(xcat, out);
}
